// LSTM_decoder_82669530514118
// MI455X (gfx1250) — compile-verified, matrix operand order checked by reading, not run on hardware
//
#include <hip/hip_runtime.h>
#include <hip/hip_bf16.h>

#define HIDDEN 1024
#define NGATES 4096
#define BATCH 256
#define SEQ_LEN 256

typedef __attribute__((ext_vector_type(16))) __bf16 v16bf;
typedef __attribute__((ext_vector_type(8)))  float  v8f;
typedef __attribute__((ext_vector_type(4)))  unsigned int u32x4;
typedef __attribute__((ext_vector_type(8)))  int          i32x8;
typedef __attribute__((ext_vector_type(4)))  int          i32x4;

union FragU {
    v16bf v;
    int4  q[2];
};

// Round-to-nearest-even f32 -> bf16 bit pattern.
__device__ __forceinline__ unsigned short f32_to_bf16_bits(float f) {
    unsigned int u = __float_as_uint(f);
    u += 0x7FFFu + ((u >> 16) & 1u);
    return (unsigned short)(u >> 16);
}

// Load a 16x32 (A) / 32x16 (B) bf16 WMMA fragment for this lane from a
// row-major row pointer (already offset by k0). CDNA5 16-bit A/B lane layout:
//   lanes 0-15 : elems 0-7 = K[0..7],  elems 8-15 = K[16..23]
//   lanes 16-31: elems 0-7 = K[8..15], elems 8-15 = K[24..31]
__device__ __forceinline__ v16bf load_frag16(const unsigned short* rowptr, int half) {
    FragU f;
    const unsigned short* p = rowptr + half * 8;   // 8 bf16 = 16 B
    f.q[0] = *(const int4*)(p);
    f.q[1] = *(const int4*)(p + 16);
    return f.v;
}

__device__ __forceinline__ float fast_sigmoid(float x) {
    return 1.0f / (1.0f + __expf(-x));
}
__device__ __forceinline__ float fast_tanh(float x) {
    float e = __expf(2.0f * x);
    return (e - 1.0f) / (e + 1.0f);
}

// ---------------------------------------------------------------------------
// TDM: DMA one 16-row x 1024-col bf16 tile (row stride 1024 elems) from
// global memory into LDS at byte offset `lds_byte_off`. D# layout per CDNA5
// ISA ch.8: group0 = {flags, lds_addr, global_addr, type}, group1 packs
// data_size / dims / strides. 2-D tensor => groups 2/3 zero.
// Dynamic LDS is the only LDS in this kernel, so its base offset is 0.
// ---------------------------------------------------------------------------
__device__ __forceinline__ void tdm_load_gate_tile(const unsigned short* gsrc,
                                                   unsigned lds_byte_off) {
    unsigned long long ga = (unsigned long long)(size_t)gsrc;
    u32x4 g0;
    g0.x = 1u;                                            // count=1, user desc
    g0.y = lds_byte_off;                                  // D#.lds_addr
    g0.z = (unsigned)(ga & 0xFFFFFFFFu);                  // global_addr[31:0]
    g0.w = (unsigned)((ga >> 32) & 0x01FFFFFFu)           // global_addr[56:32]
         | (2u << 30);                                    // type=2 ("image")
    i32x8 g1;
    g1[0] = (int)(1u << 16);        // wg_mask=0 | data_size=1 (2 bytes)
    g1[1] = (int)(1024u << 16);     // tensor_dim0[15:0]=1024  (bits 63:48)
    g1[2] = (int)(16u << 16);       // tensor_dim0[31:16]=0 | tensor_dim1[15:0]=16
    g1[3] = (int)(1024u << 16);     // tensor_dim1[31:16]=0 | tile_dim0=1024
    g1[4] = 16;                     // tile_dim1=16 | tile_dim2=0
    g1[5] = 1024;                   // tensor_dim0_stride[31:0] = 1024 elems
    g1[6] = 0;                      // stride[47:32]=0 | dim1_stride[15:0]=0
    g1[7] = 0;
    i32x4 z4 = {0, 0, 0, 0};
#if defined(__clang_major__) && (__clang_major__ >= 23)
    i32x8 z8 = {0, 0, 0, 0, 0, 0, 0, 0};
    __builtin_amdgcn_tensor_load_to_lds(g0, g1, z4, z4, z8, 0);
#else
    __builtin_amdgcn_tensor_load_to_lds(g0, g1, z4, z4, 0);
#endif
}

// ---------------------------------------------------------------------------
// Prologue: dtype conversion / packing
// ---------------------------------------------------------------------------
__global__ void f32_to_bf16_kernel(const float* __restrict__ src,
                                   unsigned short* __restrict__ dst, int n) {
    int i = blockIdx.x * blockDim.x + threadIdx.x;
    if (i < n) dst[i] = f32_to_bf16_bits(src[i]);
}

// Build Wx (4096x1024) and Wh (4096x1024) in bf16 from the four gate weights
// (each 1024x2048, cols [0,1024) = x part, [1024,2048) = h part), gate order
// i, f, o, g to match the reference concat.
__global__ void pack_gates_kernel(const float* __restrict__ Wi,
                                  const float* __restrict__ Wf,
                                  const float* __restrict__ Wo,
                                  const float* __restrict__ Wg,
                                  unsigned short* __restrict__ Wxx,
                                  unsigned short* __restrict__ Whh) {
    int row = blockIdx.x;          // 0..4095
    int g   = row >> 10;
    int j   = row & 1023;
    const float* src = (g == 0 ? Wi : g == 1 ? Wf : g == 2 ? Wo : Wg) + (size_t)j * 2048;
    for (int c = threadIdx.x; c < HIDDEN; c += blockDim.x) {
        Wxx[(size_t)row * HIDDEN + c] = f32_to_bf16_bits(src[c]);
        Whh[(size_t)row * HIDDEN + c] = f32_to_bf16_bits(src[HIDDEN + c]);
    }
}

__global__ void pack_bias_kernel(const float* __restrict__ bi,
                                 const float* __restrict__ bf,
                                 const float* __restrict__ bo,
                                 const float* __restrict__ bg,
                                 float* __restrict__ b_all) {
    int i = blockIdx.x * blockDim.x + threadIdx.x;   // 0..4095
    if (i < NGATES) {
        const float* s = (i < 1024) ? bi : (i < 2048) ? bf : (i < 3072) ? bo : bg;
        b_all[i] = s[i & 1023];
    }
}

// ---------------------------------------------------------------------------
// Generic WMMA GEMM: out = A(MxK,bf16) @ W(NxK,bf16)^T + bias
// grid = (N/16, M/128), block = 256 threads (8 waves); wave w does rows
// [blockIdx.y*128 + 16w, +16), cols [blockIdx.x*16, +16).   (prologue only)
// ---------------------------------------------------------------------------
__global__ void gemm_bias_kernel(const unsigned short* __restrict__ A,
                                 const unsigned short* __restrict__ W,
                                 const float* __restrict__ bias,
                                 float* __restrict__ outF,
                                 unsigned short* __restrict__ outB,
                                 int K, int N) {
    const int lane = threadIdx.x & 31;
    const int wave = threadIdx.x >> 5;
    const int half = lane >> 4;
    const int l15  = lane & 15;
    const int m0   = blockIdx.y * 128 + wave * 16;
    const int n0   = blockIdx.x * 16;

    const unsigned short* arow = A + (size_t)(m0 + l15) * K;
    const unsigned short* wrow = W + (size_t)(n0 + l15) * K;

    v8f acc = {};
#pragma unroll 2
    for (int k0 = 0; k0 < K; k0 += 32) {
        v16bf a = load_frag16(arow + k0, half);
        v16bf b = load_frag16(wrow + k0, half);
        acc = __builtin_amdgcn_wmma_f32_16x16x32_bf16(
            false, a, false, b, (short)0, acc, false, false);
    }

    const int col = n0 + l15;
    const float bv = bias ? bias[col] : 0.0f;
#pragma unroll
    for (int r = 0; r < 8; ++r) {
        int row = m0 + r + 8 * half;
        float v = acc[r] + bv;
        if (outF) outF[(size_t)row * N + col] = v;
        if (outB) outB[(size_t)row * N + col] = f32_to_bf16_bits(v);
    }
}

// ---------------------------------------------------------------------------
// One LSTM step, all four gates fused per wave.
//   gates = x_proj + h_in @ Whh^T ; pointwise ; write c, h(bf16), out(f32).
// grid = (1024/16, 256/128) = (64, 2), block = 256 (8 waves).
// The block's B-tile (4 gates x 16 cols = 64 W rows x 1024 K = 128 KB bf16)
// is DMA'd into LDS by the Tensor Data Mover (4 descriptors, one per gate
// tile), then all 8 waves consume it via ds_load_b128. Wave 0 issues the TDM
// ops and waits on TENSORcnt; the workgroup barrier publishes LDS to all.
// Dynamic LDS: 64*1024*2 = 131072 bytes (fits the 320 KB WGP LDS).
// ---------------------------------------------------------------------------
__global__ void lstm_step_kernel(const unsigned short* __restrict__ h_in,
                                 const unsigned short* __restrict__ Whh,
                                 const float* __restrict__ xproj,
                                 float* __restrict__ c_state,
                                 unsigned short* __restrict__ h_out,
                                 float* __restrict__ out, int t) {
    extern __shared__ unsigned short smemB[];   // [64 rows][1024 K]

    const int tid  = threadIdx.x;
    const int lane = tid & 31;
    const int wave = tid >> 5;
    const int half = lane >> 4;
    const int l15  = lane & 15;
    const int m0   = blockIdx.y * 128 + wave * 16;
    const int n0   = blockIdx.x * 16;

    // ---- TDM-stage B-tile: LDS row r (0..63) = gate (r>>4), col n0+(r&15)
    if (wave == 0) {
        for (int g = 0; g < 4; ++g) {
            const unsigned short* gsrc = Whh + (size_t)(g * HIDDEN + n0) * HIDDEN;
            tdm_load_gate_tile(gsrc, (unsigned)(g * 16 * HIDDEN * 2));
        }
        __builtin_amdgcn_s_wait_tensorcnt(0);
    }
    __syncthreads();

    const unsigned short* arow = h_in + (size_t)(m0 + l15) * HIDDEN;
    const unsigned short* b_i  = smemB + (size_t)(0 * 16 + l15) * HIDDEN;
    const unsigned short* b_f  = smemB + (size_t)(1 * 16 + l15) * HIDDEN;
    const unsigned short* b_o  = smemB + (size_t)(2 * 16 + l15) * HIDDEN;
    const unsigned short* b_g  = smemB + (size_t)(3 * 16 + l15) * HIDDEN;

    v8f ai = {}, af = {}, ao = {}, ag = {};
#pragma unroll 2
    for (int k0 = 0; k0 < HIDDEN; k0 += 32) {
        v16bf a  = load_frag16(arow + k0, half);
        v16bf fi = load_frag16(b_i + k0, half);
        v16bf ff = load_frag16(b_f + k0, half);
        v16bf fo = load_frag16(b_o + k0, half);
        v16bf fg = load_frag16(b_g + k0, half);
        ai = __builtin_amdgcn_wmma_f32_16x16x32_bf16(false, a, false, fi, (short)0, ai, false, false);
        af = __builtin_amdgcn_wmma_f32_16x16x32_bf16(false, a, false, ff, (short)0, af, false, false);
        ao = __builtin_amdgcn_wmma_f32_16x16x32_bf16(false, a, false, fo, (short)0, ao, false, false);
        ag = __builtin_amdgcn_wmma_f32_16x16x32_bf16(false, a, false, fg, (short)0, ag, false, false);
    }

    const int col = n0 + l15;
#pragma unroll
    for (int r = 0; r < 8; ++r) {
        int row = m0 + r + 8 * half;
        const float* xp = xproj + (size_t)row * NGATES;
        float gi = fast_sigmoid(ai[r] + xp[col]);
        float gf = fast_sigmoid(af[r] + xp[1024 + col]);
        float go = fast_sigmoid(ao[r] + xp[2048 + col]);
        float gg = fast_tanh  (ag[r] + xp[3072 + col]);
        size_t sidx = (size_t)row * HIDDEN + col;
        float cn = gf * c_state[sidx] + gi * gg;
        float hn = go * fast_tanh(cn);
        c_state[sidx] = cn;
        h_out[sidx]   = f32_to_bf16_bits(hn);
        // Output slab (256 MB total over the sequence) bypasses L2 residency:
        __builtin_nontemporal_store(
            hn, &out[(size_t)row * (SEQ_LEN * HIDDEN) + (size_t)t * HIDDEN + col]);
    }
}

// ---------------------------------------------------------------------------
// Host side
// ---------------------------------------------------------------------------
extern "C" void kernel_launch(void* const* d_in, const int* in_sizes, int n_in,
                              void* d_out, int out_size, void* d_ws, size_t ws_size,
                              hipStream_t stream) {
    const float* x   = (const float*)d_in[0];
    const float* c   = (const float*)d_in[1];
    const float* W_f = (const float*)d_in[2];
    const float* b_f = (const float*)d_in[3];
    const float* W_i = (const float*)d_in[4];
    const float* b_i = (const float*)d_in[5];
    const float* W_o = (const float*)d_in[6];
    const float* b_o = (const float*)d_in[7];
    const float* W_g = (const float*)d_in[8];
    const float* b_g = (const float*)d_in[9];
    const float* W_h = (const float*)d_in[10];
    const float* b_h = (const float*)d_in[11];
    const float* W_c = (const float*)d_in[12];
    const float* b_c = (const float*)d_in[13];
    float* out = (float*)d_out;

    char* ws = (char*)d_ws;
    size_t off = 0;
    auto alloc = [&](size_t bytes) -> void* {
        void* p = ws + off;
        off = (off + bytes + 255) & ~(size_t)255;
        return p;
    };
    unsigned short* Wxx    = (unsigned short*)alloc((size_t)NGATES * HIDDEN * 2);
    unsigned short* Whh    = (unsigned short*)alloc((size_t)NGATES * HIDDEN * 2);
    unsigned short* Wh0    = (unsigned short*)alloc((size_t)HIDDEN * HIDDEN * 2);
    unsigned short* Wc0    = (unsigned short*)alloc((size_t)HIDDEN * HIDDEN * 2);
    float*          b_all  = (float*)alloc((size_t)NGATES * 4);
    unsigned short* x_bf   = (unsigned short*)alloc((size_t)BATCH * HIDDEN * 2);
    unsigned short* c_bf   = (unsigned short*)alloc((size_t)BATCH * HIDDEN * 2);
    float*          xproj  = (float*)alloc((size_t)BATCH * NGATES * 4);
    unsigned short* h_buf0 = (unsigned short*)alloc((size_t)BATCH * HIDDEN * 2);
    unsigned short* h_buf1 = (unsigned short*)alloc((size_t)BATCH * HIDDEN * 2);
    float*          c_st   = (float*)alloc((size_t)BATCH * HIDDEN * 4);

    // --- prologue: conversions / packing ---
    {
        int n = BATCH * HIDDEN;
        f32_to_bf16_kernel<<<(n + 255) / 256, 256, 0, stream>>>(x, x_bf, n);
        f32_to_bf16_kernel<<<(n + 255) / 256, 256, 0, stream>>>(c, c_bf, n);
        int nw = HIDDEN * HIDDEN;
        f32_to_bf16_kernel<<<(nw + 255) / 256, 256, 0, stream>>>(W_h, Wh0, nw);
        f32_to_bf16_kernel<<<(nw + 255) / 256, 256, 0, stream>>>(W_c, Wc0, nw);
    }
    pack_gates_kernel<<<NGATES, 256, 0, stream>>>(W_i, W_f, W_o, W_g, Wxx, Whh);
    pack_bias_kernel<<<NGATES / 256, 256, 0, stream>>>(b_i, b_f, b_o, b_g, b_all);

    // --- prologue GEMMs ---
    // h0 = x @ W_h^T + b_h  -> bf16 into h_buf0
    gemm_bias_kernel<<<dim3(HIDDEN / 16, BATCH / 128), 256, 0, stream>>>(
        x_bf, Wh0, b_h, nullptr, h_buf0, HIDDEN, HIDDEN);
    // c0 = c @ W_c^T + b_c  -> f32 into c_st
    gemm_bias_kernel<<<dim3(HIDDEN / 16, BATCH / 128), 256, 0, stream>>>(
        c_bf, Wc0, b_c, c_st, nullptr, HIDDEN, HIDDEN);
    // x_proj = x @ Wx^T + b_all  -> f32 (256 x 4096)
    gemm_bias_kernel<<<dim3(NGATES / 16, BATCH / 128), 256, 0, stream>>>(
        x_bf, Wxx, b_all, xproj, nullptr, HIDDEN, NGATES);

    // --- 256 sequential steps; h double-buffered across launches ---
    const size_t stepLds = 64 * HIDDEN * sizeof(unsigned short);   // 128 KB
    for (int t = 0; t < SEQ_LEN; ++t) {
        unsigned short* h_in  = (t & 1) ? h_buf1 : h_buf0;
        unsigned short* h_out = (t & 1) ? h_buf0 : h_buf1;
        lstm_step_kernel<<<dim3(HIDDEN / 16, BATCH / 128), 256, stepLds, stream>>>(
            h_in, Whh, xproj, c_st, h_out, out, t);
    }
}
